// NERModelWithCRF_55147380081049
// MI455X (gfx1250) — matrix operand order repaired and checked by reading, not compiled, not run on hardware
//
#include <hip/hip_runtime.h>

// ---------------- problem constants ----------------
#define B_   32
#define S_   512
#define H_   768
#define NH_  12
#define L_   4
#define F_   3072
#define NL_  9
#define R_   8
#define HD_  64
#define M_TOK (B_*S_)           // 16384 tokens

// ---------------- vector types ----------------
typedef __attribute__((ext_vector_type(16))) __bf16 v16bf;
typedef __attribute__((ext_vector_type(8)))  __bf16 v8bf;
typedef __attribute__((ext_vector_type(8)))  float  v8f;
typedef __attribute__((ext_vector_type(4)))  float  v4f;

union Frag16 { v16bf v; struct { v8bf lo; v8bf hi; } p; };

__device__ __forceinline__ __bf16 f2bf(float x){ return (__bf16)x; }

__device__ __forceinline__ v8f wmma_bf16(v16bf a, v16bf b, v8f c){
  // D = A(16x32 bf16) * B(32x16 bf16) + C(16x16 f32)
  return __builtin_amdgcn_wmma_f32_16x16x32_bf16(false, a, false, b, (short)0, c, false, false);
}

// CDNA5 async copy: 16B per lane, global -> LDS, tracked on ASYNCcnt.
// lds_off = per-lane LDS byte address (low 32 bits of generic LDS pointer),
// gaddr   = per-lane 64-bit global address.
__device__ __forceinline__ void async_load_b128_to_lds(unsigned lds_off, unsigned long long gaddr){
  asm volatile("global_load_async_to_lds_b128 %0, %1, off"
               :: "v"(lds_off), "v"(gaddr) : "memory");
}
__device__ __forceinline__ void wait_asynccnt0(){
  asm volatile("s_wait_asynccnt 0" ::: "memory");
}

__device__ __forceinline__ float gelu_tanh(float x){
  float x3 = x*x*x;
  return 0.5f*x*(1.0f + tanhf(0.7978845608028654f*(x + 0.044715f*x3)));
}

// block of 256 threads (8 waves) sum reduction
__device__ __forceinline__ float blockReduceSum(float v, float* sh){
  #pragma unroll
  for(int m=16;m>=1;m>>=1) v += __shfl_xor(v, m, 32);
  int wid = threadIdx.x >> 5, lane = threadIdx.x & 31;
  if(lane==0) sh[wid] = v;
  __syncthreads();
  float t = sh[lane & 7];               // 8 waves
  #pragma unroll
  for(int m=4;m>=1;m>>=1) t += __shfl_xor(t, m, 32);
  __syncthreads();                      // sh reusable afterwards
  return t;
}

// ---------------- embedding + LN ----------------
__global__ __launch_bounds__(256)
void embed_ln_kernel(const int* __restrict__ ids, const float* __restrict__ embed,
                     const float* __restrict__ pos, const float* __restrict__ g,
                     const float* __restrict__ bta, float* __restrict__ x,
                     __bf16* __restrict__ xb){
  __shared__ float sh[8];
  int tok = blockIdx.x;
  int s = tok % S_;
  int id = ids[tok];
  const float* er = embed + (size_t)id * H_;
  const float* pr = pos + (size_t)s * H_;
  float e[3]; float sum = 0.f;
  #pragma unroll
  for(int i=0;i<3;i++){ int j = threadIdx.x + i*256; e[i] = er[j] + pr[j]; sum += e[i]; }
  float mean = blockReduceSum(sum, sh) * (1.0f/H_);
  float vs = 0.f;
  #pragma unroll
  for(int i=0;i<3;i++){ float d = e[i]-mean; vs += d*d; }
  float var = blockReduceSum(vs, sh) * (1.0f/H_);
  float rstd = rsqrtf(var + 1e-12f);
  float* xr = x + (size_t)tok*H_;
  __bf16* xbr = xb + (size_t)tok*H_;
  #pragma unroll
  for(int i=0;i<3;i++){
    int j = threadIdx.x + i*256;
    float v = (e[i]-mean)*rstd*g[j] + bta[j];
    xr[j] = v; xbr[j] = f2bf(v);
  }
}

// ---------------- residual add + LN (in-place on x) ----------------
__global__ __launch_bounds__(256)
void add_ln_kernel(float* __restrict__ x, const float* __restrict__ y,
                   const float* __restrict__ g, const float* __restrict__ bta,
                   __bf16* __restrict__ xb){
  __shared__ float sh[8];
  int tok = blockIdx.x;
  float* xr = x + (size_t)tok*H_;
  const float* yr = y + (size_t)tok*H_;
  __bf16* xbr = xb + (size_t)tok*H_;
  float e[3]; float sum = 0.f;
  #pragma unroll
  for(int i=0;i<3;i++){ int j = threadIdx.x + i*256; e[i] = xr[j] + yr[j]; sum += e[i]; }
  float mean = blockReduceSum(sum, sh) * (1.0f/H_);
  float vs = 0.f;
  #pragma unroll
  for(int i=0;i<3;i++){ float d = e[i]-mean; vs += d*d; }
  float var = blockReduceSum(vs, sh) * (1.0f/H_);
  float rstd = rsqrtf(var + 1e-12f);
  #pragma unroll
  for(int i=0;i<3;i++){
    int j = threadIdx.x + i*256;
    float v = (e[i]-mean)*rstd*g[j] + bta[j];
    xr[j] = v; xbr[j] = f2bf(v);
  }
}

// ---------------- WMMA GEMM: C[M,N] = act(A_bf16[M,K] @ W_f32[K,N] + bias) ----------------
// Block tile 64(M) x 128(N), 8 waves, wave tile 16x64 -> 4 WMMA accumulators.
// A tile staged to LDS via global_load_async_to_lds_b128 (ASYNCcnt).
// W tile converted f32->bf16 and stored transposed [n][k] so B fragments are
// contiguous ds_load_b128s.
#define LDSS 40   // padded LDS row stride in halves (80B, 16B aligned)

__global__ __launch_bounds__(256)
void gemm_bf16_wmma(const __bf16* __restrict__ A, const float* __restrict__ W,
                    const float* __restrict__ bias, float* __restrict__ Cf,
                    __bf16* __restrict__ Cb, int Ndim, int Kdim, int act){
  __shared__ __attribute__((aligned(16))) __bf16 As[64*LDSS];    // A tile 64x32
  __shared__ __attribute__((aligned(16))) __bf16 Wst[128*LDSS];  // W tile transposed: [n][k]
  const int tid = threadIdx.x;
  const int lane = tid & 31;
  const int wid = tid >> 5;
  const int wy = wid >> 1;          // 0..3 -> 16-row slice
  const int wx = wid & 1;           // 0..1 -> 64-col slice
  const int row0 = blockIdx.y * 64;
  const int col0 = blockIdx.x * 128;
  const int mlane = lane & 15;
  const int half = (lane < 16) ? 0 : 1;

  // staging coordinates
  const int arow = tid >> 2;            // 0..63
  const int ak   = (tid & 3) * 8;       // 0,8,16,24
  const int wk   = tid >> 3;            // 0..31
  const int wn   = (tid & 7) * 16;      // 0..112

  const unsigned aslot = (unsigned)(unsigned long long)(const void*)&As[arow*LDSS + ak];

  v8f acc[4] = {{},{},{},{}};

  for(int k0 = 0; k0 < Kdim; k0 += 32){
    // async copy of the A tile (no conversion needed)
    const __bf16* ap = A + (size_t)(row0 + arow)*Kdim + k0 + ak;
    async_load_b128_to_lds(aslot, (unsigned long long)ap);
    // manual staging of W tile: load f32, convert to bf16, store transposed
    const float* wp = W + (size_t)(k0 + wk)*Ndim + col0 + wn;
    v4f w0 = *(const v4f*)wp;
    v4f w1 = *(const v4f*)(wp + 4);
    v4f w2 = *(const v4f*)(wp + 8);
    v4f w3 = *(const v4f*)(wp + 12);
    if(k0 + 32 < Kdim){
      __builtin_prefetch(ap + 32, 0, 3);
      __builtin_prefetch(wp + (size_t)32*Ndim, 0, 3);
    }
    float wv[16] = {w0.x,w0.y,w0.z,w0.w, w1.x,w1.y,w1.z,w1.w,
                    w2.x,w2.y,w2.z,w2.w, w3.x,w3.y,w3.z,w3.w};
    #pragma unroll
    for(int j=0;j<16;j++) Wst[(wn+j)*LDSS + wk] = f2bf(wv[j]);
    wait_asynccnt0();                 // A tile resident in LDS
    __syncthreads();

    // A fragment (16x32): lane m = row, low lanes K 0..15, high lanes K 16..31
    Frag16 fa;
    {
      int am = wy*16 + mlane;
      int kb = half * 16;
      fa.p.lo = *(const v8bf*)&As[am*LDSS + kb];
      fa.p.hi = *(const v8bf*)&As[am*LDSS + kb + 8];
    }
    // B fragments (32x16): lane n = col, low lanes K 0..15, high lanes K 16..31
    const int kh = half * 16;
    #pragma unroll
    for(int nt=0; nt<4; nt++){
      Frag16 fb;
      int n = wx*64 + nt*16 + mlane;
      fb.p.lo = *(const v8bf*)&Wst[n*LDSS + kh];
      fb.p.hi = *(const v8bf*)&Wst[n*LDSS + kh + 8];
      acc[nt] = wmma_bf16(fa.v, fb.v, acc[nt]);
    }
    __syncthreads();
  }

  // epilogue: C layout VGPR r -> row r (low lanes) / r+8 (high lanes), lane -> col
  const int rbase = row0 + wy*16 + half*8;
  #pragma unroll
  for(int nt=0; nt<4; nt++){
    int col = col0 + wx*64 + nt*16 + mlane;
    float bvv = bias ? bias[col] : 0.f;
    #pragma unroll
    for(int r=0;r<8;r++){
      float v = acc[nt][r] + bvv;
      if(act == 1) v = gelu_tanh(v);
      size_t idx = (size_t)(rbase + r)*Ndim + col;
      if(Cf) Cf[idx] = v;
      if(Cb) Cb[idx] = f2bf(v);
    }
  }
}

// ---------------- fused attention (flash style, per (b,h), 128 q rows/block) ----------------
__global__ __launch_bounds__(256)
void attention_kernel(const __bf16* __restrict__ q, const __bf16* __restrict__ k,
                      const __bf16* __restrict__ v, const int* __restrict__ amask,
                      __bf16* __restrict__ ctx){
  __shared__ __attribute__((aligned(16))) __bf16 Vt[HD_*LDSS];        // V chunk transposed [d][key]
  __shared__ __attribute__((aligned(16))) __bf16 Plds[8*16*LDSS];     // per-wave P tile 16x32
  const int b = blockIdx.z, h = blockIdx.y;
  const int tid = threadIdx.x, lane = tid & 31, wid = tid >> 5;
  const int q0 = blockIdx.x*128 + wid*16;
  const int mlane = lane & 15;
  const int half = (lane < 16) ? 0 : 1;
  const size_t baseBH = (size_t)(b*S_)*H_ + (size_t)h*HD_;

  // Q fragments for K-dim d=0..31 and d=32..63
  Frag16 aq0, aq1;
  {
    const __bf16* qp = q + baseBH + (size_t)(q0 + mlane)*H_;
    int kb = half*16;
    aq0.p.lo = *(const v8bf*)(qp + kb);
    aq0.p.hi = *(const v8bf*)(qp + kb + 8);
    aq1.p.lo = *(const v8bf*)(qp + 32 + kb);
    aq1.p.hi = *(const v8bf*)(qp + 32 + kb + 8);
  }

  v8f o0 = {}, o1 = {}, o2 = {}, o3 = {};
  float runm[8], runs[8];
  #pragma unroll
  for(int r=0;r<8;r++){ runm[r] = -1e30f; runs[r] = 0.f; }

  __bf16* pw = &Plds[wid*16*LDSS];
  const int vkey = tid & 31;
  const int vd   = (tid >> 5) * 8;

  for(int c=0; c<S_/32; c++){
    const int s0 = c*32;
    __syncthreads();
    { // stage V chunk transposed
      const __bf16* vp = v + baseBH + (size_t)(s0 + vkey)*H_ + vd;
      v8bf vv = *(const v8bf*)vp;
      #pragma unroll
      for(int j=0;j<8;j++) Vt[(vd+j)*LDSS + vkey] = vv[j];
    }
    __syncthreads();

    // scores S = Q @ K^T for 32 keys -> two 16x16 accumulators
    v8f sc0 = {}, sc1 = {};
    {
      int kh = half*16;
      const __bf16* kp0 = k + baseBH + (size_t)(s0 + mlane)*H_;       // keys s0..s0+15
      const __bf16* kp1 = kp0 + (size_t)16*H_;                        // keys s0+16..s0+31
      Frag16 b00, b01, b10, b11;
      b00.p.lo = *(const v8bf*)(kp0 + kh);      b00.p.hi = *(const v8bf*)(kp0 + kh + 8);
      b01.p.lo = *(const v8bf*)(kp0 + 32 + kh); b01.p.hi = *(const v8bf*)(kp0 + 32 + kh + 8);
      b10.p.lo = *(const v8bf*)(kp1 + kh);      b10.p.hi = *(const v8bf*)(kp1 + kh + 8);
      b11.p.lo = *(const v8bf*)(kp1 + 32 + kh); b11.p.hi = *(const v8bf*)(kp1 + 32 + kh + 8);
      sc0 = wmma_bf16(aq0.v, b00.v, sc0);
      sc0 = wmma_bf16(aq1.v, b01.v, sc0);
      sc1 = wmma_bf16(aq0.v, b10.v, sc1);
      sc1 = wmma_bf16(aq1.v, b11.v, sc1);
    }

    // scale + mask bias (bias depends on key only)
    float bias0 = (1.f - (float)amask[b*S_ + s0 + mlane]) * -1e9f;
    float bias1 = (1.f - (float)amask[b*S_ + s0 + 16 + mlane]) * -1e9f;
    #pragma unroll
    for(int r=0;r<8;r++){
      sc0[r] = sc0[r]*0.125f + bias0;
      sc1[r] = sc1[r]*0.125f + bias1;
    }

    // online softmax (rows live in lane-halves; reduce across 16 lanes)
    #pragma unroll
    for(int r=0;r<8;r++){
      float mx = fmaxf(sc0[r], sc1[r]);
      #pragma unroll
      for(int msk=8; msk>=1; msk>>=1) mx = fmaxf(mx, __shfl_xor(mx, msk, 32));
      float nm = fmaxf(runm[r], mx);
      float corr = __expf(runm[r] - nm);
      float p0 = __expf(sc0[r] - nm);
      float p1 = __expf(sc1[r] - nm);
      float ps = p0 + p1;
      #pragma unroll
      for(int msk=8; msk>=1; msk>>=1) ps += __shfl_xor(ps, msk, 32);
      runs[r] = runs[r]*corr + ps;
      runm[r] = nm;
      o0[r] = o0[r]*corr; o1[r] = o1[r]*corr; o2[r] = o2[r]*corr; o3[r] = o3[r]*corr;
      int prow = r + half*8;
      pw[prow*LDSS + mlane]      = f2bf(p0);
      pw[prow*LDSS + 16 + mlane] = f2bf(p1);
    }

    // re-read P as A fragment (16x32)
    Frag16 ap;
    {
      int kb = half*16;
      ap.p.lo = *(const v8bf*)&pw[mlane*LDSS + kb];
      ap.p.hi = *(const v8bf*)&pw[mlane*LDSS + kb + 8];
    }
    // O += P @ V : four 16x16 WMMAs over the 64-wide head dim
    {
      int kh = half*16;
      Frag16 bv;
      int n;
      n = 0*16 + mlane; bv.p.lo = *(const v8bf*)&Vt[n*LDSS + kh]; bv.p.hi = *(const v8bf*)&Vt[n*LDSS + kh + 8];
      o0 = wmma_bf16(ap.v, bv.v, o0);
      n = 1*16 + mlane; bv.p.lo = *(const v8bf*)&Vt[n*LDSS + kh]; bv.p.hi = *(const v8bf*)&Vt[n*LDSS + kh + 8];
      o1 = wmma_bf16(ap.v, bv.v, o1);
      n = 2*16 + mlane; bv.p.lo = *(const v8bf*)&Vt[n*LDSS + kh]; bv.p.hi = *(const v8bf*)&Vt[n*LDSS + kh + 8];
      o2 = wmma_bf16(ap.v, bv.v, o2);
      n = 3*16 + mlane; bv.p.lo = *(const v8bf*)&Vt[n*LDSS + kh]; bv.p.hi = *(const v8bf*)&Vt[n*LDSS + kh + 8];
      o3 = wmma_bf16(ap.v, bv.v, o3);
    }
  }

  // normalize and write ctx (bf16, feeds Wo GEMM)
  #pragma unroll
  for(int r=0;r<8;r++){
    int row = q0 + r + half*8;
    float inv = 1.f / runs[r];
    __bf16* cp = ctx + (size_t)(b*S_ + row)*H_ + (size_t)h*HD_ + mlane;
    cp[0]  = f2bf(o0[r]*inv);
    cp[16] = f2bf(o1[r]*inv);
    cp[32] = f2bf(o2[r]*inv);
    cp[48] = f2bf(o3[r]*inv);
  }
}

// ---------------- LoRA: t[m,r] = sum_k x[m,k]*A[r,k] ----------------
__global__ __launch_bounds__(256)
void lora_xa_kernel(const float* __restrict__ x, const float* __restrict__ Aw,
                    float* __restrict__ t){
  int i = blockIdx.x*256 + threadIdx.x;          // M_TOK*R_
  int m = i >> 3, r = i & 7;
  const float* xr = x + (size_t)m*H_;
  const float* ar = Aw + (size_t)r*H_;
  float acc = 0.f;
  for(int kk=0; kk<H_; kk++) acc += xr[kk]*ar[kk];
  t[i] = acc;
}

// ---------------- LoRA add: q += scale * t @ B^T ; also emit bf16 ----------------
__global__ __launch_bounds__(256)
void lora_add_kernel(float* __restrict__ q, const float* __restrict__ t,
                     const float* __restrict__ Bw, __bf16* __restrict__ qb){
  int i = blockIdx.x*256 + threadIdx.x;          // M_TOK*H_
  int m = i / H_, n = i % H_;
  const float* tr = t + (size_t)m*R_;
  const float* br = Bw + (size_t)n*R_;
  float acc = 0.f;
  #pragma unroll
  for(int r=0;r<R_;r++) acc += tr[r]*br[r];
  float val = q[i] + 4.0f*acc;                   // LORA_SCALE = 32/8 = 4
  q[i] = val;
  qb[i] = f2bf(val);
}

// ---------------- classifier: emissions[m,c] = x @ clf_W + clf_b ----------------
__global__ __launch_bounds__(256)
void classifier_kernel(const float* __restrict__ x, const float* __restrict__ Wc,
                       const float* __restrict__ bc, float* __restrict__ em){
  int i = blockIdx.x*256 + threadIdx.x;
  if(i >= M_TOK*NL_) return;
  int m = i / NL_, c = i % NL_;
  const float* xr = x + (size_t)m*H_;
  float acc = bc[c];
  for(int kk=0; kk<H_; kk++) acc += xr[kk]*Wc[kk*NL_ + c];
  em[i] = acc;
}

// ---------------- CRF NLL per sequence ----------------
__global__ void crf_kernel(const float* __restrict__ em, const int* __restrict__ labels,
                           const int* __restrict__ amask, const float* __restrict__ startw,
                           const float* __restrict__ endw, const float* __restrict__ trans,
                           float* __restrict__ perb){
  __shared__ float tr[NL_*NL_];
  __shared__ float alpha[NL_];
  __shared__ float numsh;
  int b = blockIdx.x, j = threadIdx.x;
  const float* eb = em + (size_t)b*S_*NL_;
  const int* tg = labels + b*S_;
  const int* mk = amask + b*S_;
  for(int i=j; i<NL_*NL_; i+=32) tr[i] = trans[i];
  __syncthreads();
  if(j < NL_) alpha[j] = startw[j] + eb[j];
  if(j == 0){
    int prev = tg[0]; if(prev < 0) prev = 0;
    float num = startw[prev] + eb[prev];
    float msum = (float)mk[0];
    for(int t=1; t<S_; t++){
      int cur = tg[t]; if(cur < 0) cur = 0;
      float mf = (float)mk[t];
      num += mf * (tr[prev*NL_ + cur] + eb[t*NL_ + cur]);
      msum += mf;
      prev = cur;
    }
    int lastIdx = (int)msum - 1;
    if(lastIdx < 0) lastIdx = 0;
    if(lastIdx >= S_) lastIdx = S_-1;
    int last = tg[lastIdx]; if(last < 0) last = 0;
    num += endw[last];
    numsh = num;
  }
  __syncthreads();
  for(int t=1; t<S_; t++){
    float nx = 0.f;
    if(j < NL_){
      float mx = -1e30f;
      #pragma unroll
      for(int i2=0; i2<NL_; i2++) mx = fmaxf(mx, alpha[i2] + tr[i2*NL_ + j]);
      float sum = 0.f;
      #pragma unroll
      for(int i2=0; i2<NL_; i2++) sum += __expf(alpha[i2] + tr[i2*NL_ + j] - mx);
      nx = mx + __logf(sum) + eb[t*NL_ + j];
      if(mk[t] <= 0) nx = alpha[j];
    }
    __syncthreads();
    if(j < NL_) alpha[j] = nx;
    __syncthreads();
  }
  if(j == 0){
    float mx = -1e30f;
    for(int i2=0; i2<NL_; i2++) mx = fmaxf(mx, alpha[i2] + endw[i2]);
    float sum = 0.f;
    for(int i2=0; i2<NL_; i2++) sum += __expf(alpha[i2] + endw[i2] - mx);
    float denom = mx + __logf(sum);
    perb[b] = denom - numsh;
  }
}

__global__ void loss_reduce_kernel(const float* __restrict__ perb, float* __restrict__ out){
  float v = (threadIdx.x < B_) ? perb[threadIdx.x] : 0.f;
  #pragma unroll
  for(int m=16; m>=1; m>>=1) v += __shfl_xor(v, m, 32);
  if(threadIdx.x == 0) out[0] = v / (float)B_;   // loss = mean(denom - num)
}

// ---------------- host orchestration ----------------
extern "C" void kernel_launch(void* const* d_in, const int* in_sizes, int n_in,
                              void* d_out, int out_size, void* d_ws, size_t ws_size,
                              hipStream_t stream){
  (void)in_sizes; (void)n_in; (void)out_size; (void)ws_size;
  const int*   ids    = (const int*)d_in[0];
  const int*   amask  = (const int*)d_in[1];
  const int*   labels = (const int*)d_in[2];
  const float* embed  = (const float*)d_in[3];
  const float* pos    = (const float*)d_in[4];
  const float* lne_g  = (const float*)d_in[5];
  const float* lne_b  = (const float*)d_in[6];
  const float* Wq = (const float*)d_in[7];
  const float* bq = (const float*)d_in[8];
  const float* Wk = (const float*)d_in[9];
  const float* bk = (const float*)d_in[10];
  const float* Wv = (const float*)d_in[11];
  const float* bv = (const float*)d_in[12];
  const float* Wo = (const float*)d_in[13];
  const float* bo = (const float*)d_in[14];
  const float* Aq = (const float*)d_in[15];
  const float* Bq = (const float*)d_in[16];
  const float* Av = (const float*)d_in[17];
  const float* Bv = (const float*)d_in[18];
  const float* ln1g = (const float*)d_in[19];
  const float* ln1b = (const float*)d_in[20];
  const float* W1 = (const float*)d_in[21];
  const float* b1 = (const float*)d_in[22];
  const float* W2 = (const float*)d_in[23];
  const float* b2 = (const float*)d_in[24];
  const float* ln2g = (const float*)d_in[25];
  const float* ln2b = (const float*)d_in[26];
  const float* clfW = (const float*)d_in[27];
  const float* clfb = (const float*)d_in[28];
  const float* crf_s = (const float*)d_in[29];
  const float* crf_e = (const float*)d_in[30];
  const float* crf_t = (const float*)d_in[31];

  char* ws = (char*)d_ws;
  size_t off = 0;
  auto alloc = [&](size_t bytes)->void*{
    void* p = ws + off;
    off = (off + bytes + 255) & ~(size_t)255;
    return p;
  };
  const size_t MT = (size_t)M_TOK;
  float*  x    = (float*) alloc(MT*H_*4);
  __bf16* xb   = (__bf16*)alloc(MT*H_*2);
  float*  qf   = (float*) alloc(MT*H_*4);   // also reused as y (proj/FFN output)
  float*  vf   = (float*) alloc(MT*H_*4);
  __bf16* qbuf = (__bf16*)alloc(MT*H_*2);
  __bf16* kbuf = (__bf16*)alloc(MT*H_*2);
  __bf16* vbuf = (__bf16*)alloc(MT*H_*2);
  __bf16* ctxb = (__bf16*)alloc(MT*H_*2);
  __bf16* hb   = (__bf16*)alloc(MT*F_*2);
  float*  t1   = (float*) alloc(MT*R_*4);
  float*  t2   = (float*) alloc(MT*R_*4);
  float*  emis = (float*) alloc(MT*NL_*4);
  float*  perb = (float*) alloc((size_t)B_*4);

  dim3 blk(256);
  embed_ln_kernel<<<M_TOK, blk, 0, stream>>>(ids, embed, pos, lne_g, lne_b, x, xb);

  dim3 gproj(H_/128, M_TOK/64);    // N=768  -> 6 x 256
  dim3 gff1(F_/128, M_TOK/64);     // N=3072 -> 24 x 256
  dim3 gatt(S_/128, NH_, B_);
  const int gLoraXA  = (M_TOK*R_)/256;
  const int gLoraAdd = (M_TOK*H_)/256;

  for(int l=0; l<L_; l++){
    const float* Wq_l = Wq + (size_t)l*H_*H_;
    const float* Wk_l = Wk + (size_t)l*H_*H_;
    const float* Wv_l = Wv + (size_t)l*H_*H_;
    const float* Wo_l = Wo + (size_t)l*H_*H_;
    const float* bq_l = bq + (size_t)l*H_;
    const float* bk_l = bk + (size_t)l*H_;
    const float* bv_l = bv + (size_t)l*H_;
    const float* bo_l = bo + (size_t)l*H_;
    const float* Aq_l = Aq + (size_t)l*R_*H_;
    const float* Bq_l = Bq + (size_t)l*H_*R_;
    const float* Av_l = Av + (size_t)l*R_*H_;
    const float* Bv_l = Bv + (size_t)l*H_*R_;
    const float* W1_l = W1 + (size_t)l*H_*F_;
    const float* b1_l = b1 + (size_t)l*F_;
    const float* W2_l = W2 + (size_t)l*F_*H_;
    const float* b2_l = b2 + (size_t)l*H_;
    const float* ln1g_l = ln1g + (size_t)l*H_;
    const float* ln1b_l = ln1b + (size_t)l*H_;
    const float* ln2g_l = ln2g + (size_t)l*H_;
    const float* ln2b_l = ln2b + (size_t)l*H_;

    // Q = x@Wq + bq (+LoRA)
    gemm_bf16_wmma<<<gproj, blk, 0, stream>>>(xb, Wq_l, bq_l, qf, nullptr, H_, H_, 0);
    lora_xa_kernel<<<gLoraXA, blk, 0, stream>>>(x, Aq_l, t1);
    lora_add_kernel<<<gLoraAdd, blk, 0, stream>>>(qf, t1, Bq_l, qbuf);
    // K = x@Wk + bk  (bf16 only)
    gemm_bf16_wmma<<<gproj, blk, 0, stream>>>(xb, Wk_l, bk_l, nullptr, kbuf, H_, H_, 0);
    // V = x@Wv + bv (+LoRA)
    gemm_bf16_wmma<<<gproj, blk, 0, stream>>>(xb, Wv_l, bv_l, vf, nullptr, H_, H_, 0);
    lora_xa_kernel<<<gLoraXA, blk, 0, stream>>>(x, Av_l, t2);
    lora_add_kernel<<<gLoraAdd, blk, 0, stream>>>(vf, t2, Bv_l, vbuf);
    // attention -> ctx (bf16)
    attention_kernel<<<gatt, blk, 0, stream>>>(qbuf, kbuf, vbuf, amask, ctxb);
    // out proj -> y (reuse qf)
    gemm_bf16_wmma<<<gproj, blk, 0, stream>>>(ctxb, Wo_l, bo_l, qf, nullptr, H_, H_, 0);
    add_ln_kernel<<<M_TOK, blk, 0, stream>>>(x, qf, ln1g_l, ln1b_l, xb);
    // FFN
    gemm_bf16_wmma<<<gff1, blk, 0, stream>>>(xb, W1_l, b1_l, nullptr, hb, F_, H_, 1);
    gemm_bf16_wmma<<<gproj, blk, 0, stream>>>(hb, W2_l, b2_l, qf, nullptr, H_, F_, 0);
    add_ln_kernel<<<M_TOK, blk, 0, stream>>>(x, qf, ln2g_l, ln2b_l, xb);
  }

  classifier_kernel<<<(M_TOK*NL_ + 255)/256, blk, 0, stream>>>(x, clfW, clfb, emis);
  crf_kernel<<<B_, 32, 0, stream>>>(emis, labels, amask, crf_s, crf_e, crf_t, perb);
  loss_reduce_kernel<<<1, 32, 0, stream>>>(perb, (float*)d_out);
}
